// GCN_Sparse_Memory_Policy_SelectNode_10_66726611911068
// MI455X (gfx1250) — matrix operand order repaired and checked by reading, not compiled, not run on hardware
//
#include <hip/hip_runtime.h>
#include <hip/hip_bf16.h>
#include <math.h>

// ---------------------------------------------------------------------------
// GCN: 9 x [ relu( spmm(adj, x @ W) + b ) ] ; then 128->1, spmm, log_softmax
//   N = 100000 nodes, E = 1600000 edges, d = 128
// GEMMs run on v_wmma_f32_16x16x32_bf16 (f32 accumulate); W staged into LDS by
// the Tensor Data Mover (tensor_load_to_lds + s_wait_tensorcnt); SpMM is
// wave-per-edge with hardware global_atomic_add_f32 into an L2-resident
// accumulator.
// ---------------------------------------------------------------------------

#define N_NODES 100000
#define NH      128
#define N_MID   9

typedef __attribute__((ext_vector_type(8)))  float  v8f;
typedef __attribute__((ext_vector_type(16))) __bf16 v16bf;
typedef __attribute__((ext_vector_type(8)))  __bf16 v8bf;

typedef __attribute__((ext_vector_type(4))) unsigned int ui32x4;
typedef __attribute__((ext_vector_type(8))) int          i32x8;
typedef __attribute__((ext_vector_type(4))) int          i32x4;

static __device__ __forceinline__ v16bf pack16(v8bf lo, v8bf hi) {
    v16bf r;
#pragma unroll
    for (int i = 0; i < 8; ++i) { r[i] = lo[i]; r[i + 8] = hi[i]; }
    return r;
}

// ---------------- f32 -> bf16 conversion (features -> xb) -------------------
__global__ void k_cvt_bf16(const float* __restrict__ x, __bf16* __restrict__ xb, int n) {
    int i = blockIdx.x * blockDim.x + threadIdx.x;
    if (i < n) xb[i] = (__bf16)x[i];
}

// ------------- Ws[l][k][n] (f32) -> Wt[l][n][k] (bf16, transposed) ----------
__global__ void k_twt(const float* __restrict__ Ws, __bf16* __restrict__ Wt, int total) {
    int i = blockIdx.x * blockDim.x + threadIdx.x;
    if (i >= total) return;
    int l = i >> 14, r = i & 16383, n = r >> 7, k = r & 127;
    Wt[i] = (__bf16)Ws[(l << 14) + (k << 7) + n];   // Wt[l][n][k] = Ws[l][k][n]
}

// ---------------- WMMA GEMM: support[M x 128] = xb[M x 128] @ W -------------
// W held transposed (Wt[n][k], bf16); TDM DMAs the 32 KB tile into LDS.
// One wave computes a 16(M) x 128(N) stripe: 4 K-steps x 8 N-tiles = 32 WMMAs,
// A-fragment reused across the 8 N-tiles of each K-step.
__global__ void __launch_bounds__(256, 2)
k_gemm_wmma(const __bf16* __restrict__ xb, const __bf16* __restrict__ Wt,
            float* __restrict__ support, int M) {
    __shared__ __bf16 lds[128 * 128];     // 32 KB (of 320 KB per WGP)

#if __has_builtin(__builtin_amdgcn_tensor_load_to_lds)
    if (threadIdx.x < 32) {
        // Tensor DMA descriptor (ISA ch.8): 2D tile, 128 x 128 x 2B, stride 128.
        const unsigned long long gaddr = (unsigned long long)(const void*)Wt;
        const unsigned int lds_off = (unsigned int)(unsigned long long)(const void*)&lds[0];
        ui32x4 g0;
        g0[0] = 1u;                                            // count=1 (valid D#)
        g0[1] = lds_off;                                       // lds_addr
        g0[2] = (unsigned int)(gaddr & 0xffffffffu);           // global_addr[31:0]
        g0[3] = (unsigned int)((gaddr >> 32) & 0x01ffffffu)    // global_addr[56:32]
              | (2u << 30);                                    // type=2 ("image")
        i32x8 g1;
        g1[0] = (int)(1u << 16);            // wg_mask=0, data_size=1 (2 bytes)
        g1[1] = (int)(128u << 16);          // tensor_dim0[15:0]=128
        g1[2] = (int)(128u << 16);          // tensor_dim0 hi=0 | tensor_dim1 lo=128
        g1[3] = (int)(128u << 16);          // tensor_dim1 hi=0 | tile_dim0=128
        g1[4] = (int)128u;                  // tile_dim1=128, tile_dim2=0
        g1[5] = (int)128u;                  // tensor_dim0_stride=128 (elements)
        g1[6] = 0;                          // stride0 hi | tensor_dim1_stride lo
        g1[7] = 0;                          // tensor_dim1_stride hi
        i32x4 gz = {0, 0, 0, 0};            // groups 2/3: 2-D tensor -> unused
#if __clang_major__ >= 23
        i32x8 gz8 = {0, 0, 0, 0, 0, 0, 0, 0};
        __builtin_amdgcn_tensor_load_to_lds(g0, g1, gz, gz, gz8, 0);
#else
        __builtin_amdgcn_tensor_load_to_lds(g0, g1, gz, gz, 0);
#endif
        __builtin_amdgcn_s_wait_tensorcnt(0);
    }
#else
    {   // fallback: cooperative 32KB copy, 256 threads x 8 x 16B
        const uint4* src = (const uint4*)Wt;
        uint4*       dst = (uint4*)lds;
#pragma unroll
        for (int i = 0; i < 8; ++i) dst[threadIdx.x + i * 256] = src[threadIdx.x + i * 256];
    }
#endif
    __syncthreads();

    const int t     = threadIdx.x;
    const int wave  = t >> 5;
    const int lane  = t & 31;
    const int mbase = (blockIdx.x * 8 + wave) * 16;
    if (mbase >= M) return;                 // wave-uniform: EXEC stays all-ones

    const int half = lane >> 4;             // 0 or 1
    const int klo  = half << 3;             // unused: A also uses kbB (K sub-offset 0 / 16)
    const int kbB  = half << 4;             // B: K sub-offset 0 / 16
    const int ncol = lane & 15;
    const int mrow = mbase + ncol;          // A: lane = M (lane & 15)

    v8f acc[8] = {};
    const __bf16* arow = xb + (size_t)mrow * 128;

#pragma unroll
    for (int ks = 0; ks < 4; ++ks) {
        const __bf16* ap = arow + ks * 32;
        v16bf afrag = pack16(*(const v8bf*)(ap + kbB),
                             *(const v8bf*)(ap + kbB + 8));
#pragma unroll
        for (int nt = 0; nt < 8; ++nt) {
            const __bf16* bp = &lds[(nt * 16 + ncol) * 128 + ks * 32 + kbB];
            v16bf bfrag = pack16(((const v8bf*)bp)[0], ((const v8bf*)bp)[1]);
            acc[nt] = __builtin_amdgcn_wmma_f32_16x16x32_bf16(
                false, afrag, false, bfrag, (short)0, acc[nt], false, false);
        }
    }

    // D layout: VGPR r -> row mbase + r + 8*half, col = nt*16 + (lane&15)
    const int rbase = mbase + (half << 3);
#pragma unroll
    for (int nt = 0; nt < 8; ++nt)
#pragma unroll
        for (int r = 0; r < 8; ++r)
            support[(size_t)(rbase + r) * 128 + nt * 16 + ncol] = acc[nt][r];
}

// ---------------- y[n][k] = b[k] (bias pre-load of accumulator) -------------
__global__ void k_bias_init(float* __restrict__ y, const float* __restrict__ b, int n) {
    int i = blockIdx.x * blockDim.x + threadIdx.x;
    if (i < n) y[i] = b[i & 127];
}

// ---------------- SpMM scatter: y[dst] += val * support[src] ----------------
// Wave-per-edge: 32 lanes x float4 = one coalesced 512B row gather,
// 128 hardware f32 atomics into the L2-resident accumulator.
__global__ void k_spmm(const float* __restrict__ support,
                       const int* __restrict__ esrc, const int* __restrict__ edst,
                       const float* __restrict__ ev, float* __restrict__ y, int E) {
    const int lane   = threadIdx.x & 31;
    const int wave   = (blockIdx.x * blockDim.x + threadIdx.x) >> 5;
    const int nwaves = (gridDim.x * blockDim.x) >> 5;
    for (int e = wave; e < E; e += nwaves) {
        if (e + nwaves < E) {               // global_prefetch_b8 for next record
            __builtin_prefetch(&esrc[e + nwaves], 0, 0);
            __builtin_prefetch(&edst[e + nwaves], 0, 0);
        }
        const int   s = esrc[e];
        const int   d = edst[e];
        const float v = ev[e];
        const float4 xr = *(const float4*)(support + (size_t)s * 128 + lane * 4);
        float* yp = y + (size_t)d * 128 + lane * 4;
        __hip_atomic_fetch_add(yp + 0, v * xr.x, __ATOMIC_RELAXED, __HIP_MEMORY_SCOPE_AGENT);
        __hip_atomic_fetch_add(yp + 1, v * xr.y, __ATOMIC_RELAXED, __HIP_MEMORY_SCOPE_AGENT);
        __hip_atomic_fetch_add(yp + 2, v * xr.z, __ATOMIC_RELAXED, __HIP_MEMORY_SCOPE_AGENT);
        __hip_atomic_fetch_add(yp + 3, v * xr.w, __ATOMIC_RELAXED, __HIP_MEMORY_SCOPE_AGENT);
    }
}

// ---------------- fused ReLU + f32->bf16 (feeds next GEMM) ------------------
__global__ void k_relu_bf16(const float* __restrict__ y, __bf16* __restrict__ xb, int n) {
    int i = blockIdx.x * blockDim.x + threadIdx.x;
    if (i < n) xb[i] = (__bf16)fmaxf(y[i], 0.0f);
}

// ---------------- layer 10: z[n] = xb[n] . W10  (wave shuffle reduce) -------
__global__ void k_dot128(const __bf16* __restrict__ xb, const float* __restrict__ W10,
                         float* __restrict__ z, int N) {
    const int lane   = threadIdx.x & 31;
    const int wave   = (blockIdx.x * blockDim.x + threadIdx.x) >> 5;
    const int nwaves = (gridDim.x * blockDim.x) >> 5;
    for (int n = wave; n < N; n += nwaves) {
        const __bf16* xr = xb + (size_t)n * 128 + lane * 4;
        float s = (float)xr[0] * W10[lane * 4 + 0] + (float)xr[1] * W10[lane * 4 + 1] +
                  (float)xr[2] * W10[lane * 4 + 2] + (float)xr[3] * W10[lane * 4 + 3];
#pragma unroll
        for (int off = 16; off; off >>= 1) s += __shfl_xor(s, off, 32);
        if (lane == 0) z[n] = s;
    }
}

__global__ void k_scalar_init(float* __restrict__ o, const float* __restrict__ b10, int n) {
    int i = blockIdx.x * blockDim.x + threadIdx.x;
    if (i < n) o[i] = b10[0];
}

__global__ void k_scatter1(const float* __restrict__ z,
                           const int* __restrict__ esrc, const int* __restrict__ edst,
                           const float* __restrict__ ev, float* __restrict__ o, int E) {
    for (int e = blockIdx.x * blockDim.x + threadIdx.x; e < E; e += gridDim.x * blockDim.x)
        __hip_atomic_fetch_add(&o[edst[e]], ev[e] * z[esrc[e]],
                               __ATOMIC_RELAXED, __HIP_MEMORY_SCOPE_AGENT);
}

// ---------------- log_softmax: two-stage max / sum-exp reductions -----------
__global__ void k_max_part(const float* __restrict__ x, int n, float* __restrict__ pmax) {
    __shared__ float sm[256];
    float m = -__builtin_inff();
    for (int i = blockIdx.x * blockDim.x + threadIdx.x; i < n; i += gridDim.x * blockDim.x)
        m = fmaxf(m, x[i]);
    sm[threadIdx.x] = m;
    __syncthreads();
    for (int s = 128; s; s >>= 1) {
        if ((int)threadIdx.x < s) sm[threadIdx.x] = fmaxf(sm[threadIdx.x], sm[threadIdx.x + s]);
        __syncthreads();
    }
    if (!threadIdx.x) pmax[blockIdx.x] = sm[0];
}

__global__ void k_max_final(const float* __restrict__ pmax, float* __restrict__ gstats) {
    __shared__ float sm[256];
    sm[threadIdx.x] = pmax[threadIdx.x];
    __syncthreads();
    for (int s = 128; s; s >>= 1) {
        if ((int)threadIdx.x < s) sm[threadIdx.x] = fmaxf(sm[threadIdx.x], sm[threadIdx.x + s]);
        __syncthreads();
    }
    if (!threadIdx.x) gstats[0] = sm[0];
}

__global__ void k_sumexp_part(const float* __restrict__ x, int n,
                              const float* __restrict__ gstats, float* __restrict__ psum) {
    __shared__ float sm[256];
    const float gmax = gstats[0];
    float acc = 0.0f;
    for (int i = blockIdx.x * blockDim.x + threadIdx.x; i < n; i += gridDim.x * blockDim.x)
        acc += expf(x[i] - gmax);
    sm[threadIdx.x] = acc;
    __syncthreads();
    for (int s = 128; s; s >>= 1) {
        if ((int)threadIdx.x < s) sm[threadIdx.x] += sm[threadIdx.x + s];
        __syncthreads();
    }
    if (!threadIdx.x) psum[blockIdx.x] = sm[0];
}

__global__ void k_sumexp_final(const float* __restrict__ psum, float* __restrict__ gstats) {
    __shared__ float sm[256];
    sm[threadIdx.x] = psum[threadIdx.x];
    __syncthreads();
    for (int s = 128; s; s >>= 1) {
        if ((int)threadIdx.x < s) sm[threadIdx.x] += sm[threadIdx.x + s];
        __syncthreads();
    }
    if (!threadIdx.x) gstats[1] = logf(sm[0]);
}

__global__ void k_final(const float* __restrict__ x, const float* __restrict__ gstats,
                        float* __restrict__ out, int n) {
    int i = blockIdx.x * blockDim.x + threadIdx.x;
    if (i < n) out[i] = x[i] - gstats[0] - gstats[1];
}

// ---------------------------------------------------------------------------
extern "C" void kernel_launch(void* const* d_in, const int* in_sizes, int n_in,
                              void* d_out, int out_size, void* d_ws, size_t ws_size,
                              hipStream_t stream) {
    const float* features = (const float*)d_in[0];   // [N,128]
    const float* adj_vals = (const float*)d_in[1];   // [E]
    const float* Ws       = (const float*)d_in[2];   // [9,128,128]
    const float* bs       = (const float*)d_in[3];   // [9,128]
    const float* W10      = (const float*)d_in[4];   // [128,1]
    const float* b10      = (const float*)d_in[5];   // [1]
    const int*   esrc     = (const int*)d_in[6];     // [E]
    const int*   edst     = (const int*)d_in[7];     // [E]
    const int    E        = in_sizes[6];
    float*       out      = (float*)d_out;

    // workspace layout (256B-aligned regions)
    char* ws = (char*)d_ws;
    __bf16* xb      = (__bf16*)(ws + 0);                      // 25,600,000 B
    float*  support = (float*) (ws + 25600000);               // 51,200,000 B
    float*  y       = (float*) (ws + 76800000);               // 51,200,000 B
    __bf16* Wt_all  = (__bf16*)(ws + 128000000);              //    294,912 B
    float*  z       = (float*) (ws + 128294912);              //    400,000 B
    float*  o       = (float*) (ws + 128695040);              //    400,000 B
    float*  pmax    = (float*) (ws + 129095168);              //      1,024 B
    float*  psum    = (float*) (ws + 129096192);              //      1,024 B
    float*  gstats  = (float*) (ws + 129097216);              //          8 B

    const int NE128 = N_NODES * NH;          // 12,800,000
    const int BLK   = 256;

    // one-time per call: bf16 activations + transposed bf16 weights
    k_cvt_bf16<<<(NE128 + BLK - 1) / BLK, BLK, 0, stream>>>(features, xb, NE128);
    k_twt<<<(N_MID * NH * NH + BLK - 1) / BLK, BLK, 0, stream>>>(Ws, Wt_all, N_MID * NH * NH);

    const int gemm_blocks = (N_NODES / 16 + 7) / 8;   // 782
    for (int l = 0; l < N_MID; ++l) {
        k_gemm_wmma<<<gemm_blocks, BLK, 0, stream>>>(xb, Wt_all + l * NH * NH, support, N_NODES);
        k_bias_init<<<(NE128 + BLK - 1) / BLK, BLK, 0, stream>>>(y, bs + l * NH, NE128);
        k_spmm<<<2048, BLK, 0, stream>>>(support, esrc, edst, adj_vals, y, E);
        k_relu_bf16<<<(NE128 + BLK - 1) / BLK, BLK, 0, stream>>>(y, xb, NE128);
    }

    // layer 10: 128 -> 1, scatter, log_softmax
    k_dot128<<<12500, BLK, 0, stream>>>(xb, W10, z, N_NODES);
    k_scalar_init<<<(N_NODES + BLK - 1) / BLK, BLK, 0, stream>>>(o, b10, N_NODES);
    k_scatter1<<<(E + BLK - 1) / BLK, BLK, 0, stream>>>(z, esrc, edst, adj_vals, o, E);

    k_max_part<<<256, BLK, 0, stream>>>(o, N_NODES, pmax);
    k_max_final<<<1, BLK, 0, stream>>>(pmax, gstats);
    k_sumexp_part<<<256, BLK, 0, stream>>>(o, N_NODES, gstats, psum);
    k_sumexp_final<<<1, BLK, 0, stream>>>(psum, gstats);
    k_final<<<(N_NODES + BLK - 1) / BLK, BLK, 0, stream>>>(o, gstats, out, N_NODES);
}
